// LSTMAutoencoder_59992103190879
// MI455X (gfx1250) — matrix operand order repaired and checked by reading, not compiled, not run on hardware
//
#include <hip/hip_runtime.h>
#include <hip/hip_bf16.h>

// Problem constants (match reference)
#define BB  256   // batch
#define TT  512   // seq len
#define II  128   // input size
#define HH  128   // hidden size
#define G4H 512   // 4*H

// Persistent grid: 4 batch-groups x 8 hidden-slices
#define NWG  32
#define NTHR 128  // 4 wave32s

typedef __bf16 bf16;
typedef __attribute__((ext_vector_type(8)))  __bf16 v8bf;
typedef __attribute__((ext_vector_type(16))) __bf16 v16bf;
typedef __attribute__((ext_vector_type(8)))  float  v8f;

// ---- workspace layout (global scratch d_ws) ----
// [0,256)                : barrier counter (u32) + pad
// h buffers: h[layer][parity] : 4 x (B*H) bf16
// y buffers: y[parity]        : 2 x (B*I) bf16
#define OFF_H      256
#define HBUF_ELEMS ((size_t)BB * HH)
#define OFF_Y      (OFF_H + 4 * HBUF_ELEMS * sizeof(bf16))
#define YBUF_ELEMS ((size_t)BB * II)

__device__ __forceinline__ float sigm(float x)   { return 1.0f / (1.0f + __expf(-x)); }
__device__ __forceinline__ float tanh_e(float x) { return 2.0f / (1.0f + __expf(-2.0f * x)) - 1.0f; }

// Device-scope barrier: every WG adds 1 per barrier; after k barriers the
// counter equals k*NWG. Counter is zeroed by the init kernel each call.
__device__ __forceinline__ void gbarrier(unsigned* cnt, unsigned target) {
    __threadfence();
    __syncthreads();
    if (threadIdx.x == 0) {
        __hip_atomic_fetch_add(cnt, 1u, __ATOMIC_ACQ_REL, __HIP_MEMORY_SCOPE_AGENT);
        while (__hip_atomic_load(cnt, __ATOMIC_ACQUIRE, __HIP_MEMORY_SCOPE_AGENT) < target)
            __builtin_amdgcn_s_sleep(1);
    }
    __syncthreads();
}

// One K=128 GEMM contribution into NG 16x16 accumulators.
// A: activation rows (global), per CDNA5 16-bit A layout:
//   lane<16 holds row (lane&15), K {k0..k0+7} in v0-3 and {k0+16..k0+23} in v4-7;
//   lane>=16 same row, K shifted by +8 (handled via half*8).
// B: LDS weight rows [NG*16][128] (W[n][k], row-major), same K order as A:
//   lane n holds col (lane&15), K {k0..k0+7} and {k0+16..k0+23}, shifted by +8 for lane>=16.
template<int NG, bool AF32>
__device__ __forceinline__ void mm_src(v8f* acc, const void* __restrict__ Ain, int sA, int rowA0,
                                       const bf16* __restrict__ B, int lr, int half) {
#pragma unroll
    for (int kt = 0; kt < 4; ++kt) {
        const int k0 = kt * 32;
        v16bf a;
        if constexpr (AF32) {
            const float* ap = (const float*)Ain + (size_t)(rowA0 + lr) * sA + half * 8 + k0;
            v8f flo = *(const v8f*)(ap);
            v8f fhi = *(const v8f*)(ap + 16);
#pragma unroll
            for (int i = 0; i < 8; ++i) { a[i] = (bf16)flo[i]; a[i + 8] = (bf16)fhi[i]; }
        } else {
            const bf16* ap = (const bf16*)Ain + (size_t)(rowA0 + lr) * sA + half * 8 + k0;
            v8bf alo = *(const v8bf*)(ap);
            v8bf ahi = *(const v8bf*)(ap + 16);
            a = __builtin_shufflevector(alo, ahi, 0,1,2,3,4,5,6,7,8,9,10,11,12,13,14,15);
        }
#pragma unroll
        for (int g = 0; g < NG; ++g) {
            const bf16* bp = B + (size_t)(g * 16 + lr) * 128 + k0 + half * 8;
            v16bf b = __builtin_shufflevector(*(const v8bf*)(bp), *(const v8bf*)(bp + 16), 0,1,2,3,4,5,6,7,8,9,10,11,12,13,14,15);
            acc[g] = __builtin_amdgcn_wmma_f32_16x16x32_bf16(false, a, false, b,
                                                             (short)0, acc[g], false, false);
        }
    }
}

// One LSTM layer phase for this wave's 16 batch rows x this WG's 16 hidden units.
// wl = LDS weights [2(src)][64(gate rows i,f,g,o x16)][128]; gates land in acc[0..3]
// for the SAME (batch,unit) element per lane -> local elementwise c/h update.
template<bool AF32>
__device__ __forceinline__ void lstm_layer(
    const void* __restrict__ Ain, int sAin,
    const bf16* __restrict__ Hprev, bf16* __restrict__ Hout,
    const bf16* __restrict__ wl, const float* __restrict__ b4,
    float* __restrict__ cSl,
    int rowB0, int mL0, int outCol, int lr, int half) {
    v8f acc[4] = {};
    mm_src<4, AF32 >(acc, Ain, sAin, rowB0, wl, lr, half);
    mm_src<4, false>(acc, (const void*)Hprev, HH, rowB0, wl + 64 * 128, lr, half);
#pragma unroll
    for (int r = 0; r < 8; ++r) {
        float gi = sigm  (acc[0][r] + b4[0]);
        float gf = sigm  (acc[1][r] + b4[1]);
        float gg = tanh_e(acc[2][r] + b4[2]);
        float go = sigm  (acc[3][r] + b4[3]);
        int   mloc = mL0 + r + half * 8;          // row within this WG's 64-row batch group
        float* cp  = cSl + lr * 64 + mloc;
        float cn   = gf * cp[0] + gi * gg;
        cp[0] = cn;
        Hout[(size_t)(rowB0 + r + half * 8) * HH + outCol] = (bf16)(go * tanh_e(cn));
    }
}

// Init: zero h ping-pong buffers, broadcast start token into y buffers, zero barrier.
__global__ void init_ws(const float* __restrict__ stok, char* __restrict__ ws) {
    size_t i = (size_t)blockIdx.x * blockDim.x + threadIdx.x;
    bf16* hb = (bf16*)(ws + OFF_H);
    if (i < 4 * HBUF_ELEMS) hb[i] = (bf16)0.0f;
    bf16* yb = (bf16*)(ws + OFF_Y);
    if (i < 2 * YBUF_ELEMS) yb[i] = (bf16)stok[i % II];
    if (i == 0) *(unsigned*)ws = 0u;
}

__global__ void __launch_bounds__(NTHR, 1) lstm_ae_persistent(
    const float* __restrict__ x,
    const float* __restrict__ eWih, const float* __restrict__ eWhh, const float* __restrict__ eb,
    const float* __restrict__ dWih, const float* __restrict__ dWhh, const float* __restrict__ db,
    const float* __restrict__ Wout, const float* __restrict__ bout,
    float* __restrict__ out, char* __restrict__ ws) {
    // Dynamic LDS carve: 8KB f32 cell state + 128KB bf16 enc/dec weights + 4KB Wout slice
    extern __shared__ char smem[];
    float* cSt  = (float*)smem;                      // [2 layers][16 units][64 rows]
    bf16* wEnc  = (bf16*)(smem + 8192);              // [2 layers][2 src][64][128]
    bf16* wDec  = wEnc + 2 * 2 * 64 * 128;
    bf16* wOutL = wDec + 2 * 2 * 64 * 128;           // [16][128]

    const int tid  = threadIdx.x;
    const int lane = tid & 31, w = tid >> 5;
    const int lr   = lane & 15, half = lane >> 4;
    const int bg   = blockIdx.x >> 3;                // batch group 0..3
    const int hs   = blockIdx.x & 7;                 // hidden slice 0..7
    const int rowB0  = bg * 64 + w * 16;             // this wave's batch rows
    const int mL0    = w * 16;                       // local row base inside WG
    const int outCol = hs * 16 + lr;                 // hidden unit / output column

    // ---- one-time: gather+convert weight slices (gate rows i,f,g,o for our 16 units) ----
    for (int idx = tid; idx < 2 * 2 * 64 * 128; idx += NTHR) {
        int l = idx >> 14, s = (idx >> 13) & 1, r = (idx >> 7) & 63, k = idx & 127;
        int grow = (r >> 4) * HH + hs * 16 + (r & 15);          // gate*128 + unit
        size_t gi = (size_t)l * G4H * II + (size_t)grow * II + k;
        wEnc[idx] = (bf16)((s ? eWhh : eWih)[gi]);
        wDec[idx] = (bf16)((s ? dWhh : dWih)[gi]);
    }
    for (int idx = tid; idx < 16 * 128; idx += NTHR)
        wOutL[idx] = (bf16)Wout[(size_t)(hs * 16 + (idx >> 7)) * HH + (idx & 127)];
    for (int idx = tid; idx < 2 * 16 * 64; idx += NTHR) cSt[idx] = 0.0f;
    __syncthreads();

    // per-lane gate biases
    float ebv[2][4], dbv[2][4];
#pragma unroll
    for (int l = 0; l < 2; ++l)
#pragma unroll
        for (int g = 0; g < 4; ++g) {
            ebv[l][g] = eb[l * G4H + g * HH + outCol];
            dbv[l][g] = db[l * G4H + g * HH + outCol];
        }
    const float boutv = bout[outCol];

    bf16* hB = (bf16*)(ws + OFF_H);
    bf16* yB = (bf16*)(ws + OFF_Y);
    unsigned* cnt = (unsigned*)ws;
    unsigned bar = 0;

    // =========================== ENCODER ===========================
#pragma clang loop unroll(disable)
    for (int t = 0; t < TT; ++t) {
        const int cur = t & 1, prv = cur ^ 1;
        bf16* h0c = hB + (0 * 2 + cur) * HBUF_ELEMS;
        bf16* h0p = hB + (0 * 2 + prv) * HBUF_ELEMS;
        bf16* h1c = hB + (1 * 2 + cur) * HBUF_ELEMS;
        bf16* h1p = hB + (1 * 2 + prv) * HBUF_ELEMS;

        if (t + 1 < TT)
            __builtin_prefetch(x + (size_t)(rowB0 + lr) * TT * II + (size_t)(t + 1) * II, 0, 1);

        // layer 0: A = x_t (fp32 -> bf16 on the fly), stride T*I
        lstm_layer<true>(x + (size_t)t * II, TT * II, h0p, h0c,
                         wEnc, ebv[0], cSt, rowB0, mL0, outCol, lr, half);
        bar++; gbarrier(cnt, bar * NWG);
        // layer 1: A = h0_t
        lstm_layer<false>(h0c, HH, h1p, h1c,
                          wEnc + 2 * 64 * 128, ebv[1], cSt + 16 * 64,
                          rowB0, mL0, outCol, lr, half);
        bar++; gbarrier(cnt, bar * NWG);
    }

    // =========================== DECODER ===========================
    // Parity continues: t=0 reads index 1 = encoder finals / start token.
#pragma clang loop unroll(disable)
    for (int t = 0; t < TT; ++t) {
        const int cur = t & 1, prv = cur ^ 1;
        bf16* h0c = hB + (0 * 2 + cur) * HBUF_ELEMS;
        bf16* h0p = hB + (0 * 2 + prv) * HBUF_ELEMS;
        bf16* h1c = hB + (1 * 2 + cur) * HBUF_ELEMS;
        bf16* h1p = hB + (1 * 2 + prv) * HBUF_ELEMS;
        bf16* yc  = yB + cur * YBUF_ELEMS;
        bf16* yp  = yB + prv * YBUF_ELEMS;

        lstm_layer<false>(yp, II, h0p, h0c,
                          wDec, dbv[0], cSt, rowB0, mL0, outCol, lr, half);
        bar++; gbarrier(cnt, bar * NWG);
        lstm_layer<false>(h0c, HH, h1p, h1c,
                          wDec + 2 * 64 * 128, dbv[1], cSt + 16 * 64,
                          rowB0, mL0, outCol, lr, half);
        bar++; gbarrier(cnt, bar * NWG);

        // projection: y_t = h1_t @ Wout^T + bout ; emit fp32 output + bf16 feedback
        {
            v8f pacc[1] = {};
            mm_src<1, false>(pacc, (const void*)h1c, HH, rowB0, wOutL, lr, half);
#pragma unroll
            for (int r = 0; r < 8; ++r) {
                float v = pacc[0][r] + boutv;
                int   m = rowB0 + r + half * 8;
                out[(size_t)m * TT * II + (size_t)t * II + outCol] = v;
                yc[(size_t)m * II + outCol] = (bf16)v;
            }
        }
        bar++; gbarrier(cnt, bar * NWG);
    }
}

extern "C" void kernel_launch(void* const* d_in, const int* in_sizes, int n_in,
                              void* d_out, int out_size, void* d_ws, size_t ws_size,
                              hipStream_t stream) {
    const float* x    = (const float*)d_in[0];
    const float* eWih = (const float*)d_in[1];
    const float* eWhh = (const float*)d_in[2];
    const float* eb   = (const float*)d_in[3];
    const float* dWih = (const float*)d_in[4];
    const float* dWhh = (const float*)d_in[5];
    const float* db   = (const float*)d_in[6];
    const float* Wout = (const float*)d_in[7];
    const float* bout = (const float*)d_in[8];
    const float* stok = (const float*)d_in[9];
    char*  ws  = (char*)d_ws;
    float* out = (float*)d_out;

    // Init h/y/barrier state (must cover 4*B*H elements)
    size_t n_init = 4 * HBUF_ELEMS;
    int blocks = (int)((n_init + 255) / 256);
    init_ws<<<blocks, 256, 0, stream>>>(stok, ws);

    // 8KB cell state + 128KB weights + 4KB Wout slice = 140KB dynamic LDS / WG
    size_t shmem = 8192 + (size_t)2 * (2 * 2 * 64 * 128 * sizeof(bf16)) + 16 * 128 * sizeof(bf16);
    lstm_ae_persistent<<<NWG, NTHR, shmem, stream>>>(x, eWih, eWhh, eb, dWih, dWhh, db,
                                                     Wout, bout, out, ws);
}
